// FeatureCorrelator_39238821216846
// MI455X (gfx1250) — matrix operand order repaired and checked by reading, not compiled, not run on hardware
//
#include <hip/hip_runtime.h>
#include <hip/hip_bf16.h>

// ---------------------------------------------------------------------------
// FeatureCorrelator for MI455X (gfx1250, wave32, WMMA).
// B=1, N=4096, D=512, K=16.  Main cost: MLP GEMMs -> bf16 WMMA 16x16x32.
// Algebraic fold: new_pts = [g1, g2, a*(g2-g1)/s + b] hits disjoint w0 slices,
// so fold a/s into effective weights; the g1 part depends only on n and is
// precomputed once per point as a row-bias for the 65536-row g2 GEMM.
// GEMM staging uses CDNA5 async global->LDS DMA (ASYNCcnt) double-buffered
// so the k+1 tile transfer overlaps the k WMMA work.
// ---------------------------------------------------------------------------

#define NPTS 4096
#define DCH  512
#define K_NN 16
#define CCH  515               // D+3
#define KP   544               // 515 padded to multiple of 32
#define MROWS (NPTS * K_NN)    // 65536
#define NOUT 512

typedef __attribute__((ext_vector_type(16))) __bf16 v16bf;
typedef __attribute__((ext_vector_type(8)))  float  v8f;

union FragAB { unsigned int u[8]; v16bf v; };

// ---------------------------------------------------------------- KNN ------
__global__ void knn_kernel(const float* __restrict__ x1, const float* __restrict__ x2,
                           int* __restrict__ knn_idx) {
  int n = blockIdx.x * blockDim.x + threadIdx.x;
  if (n >= NPTS) return;
  float ax = x1[0 * NPTS + n], ay = x1[1 * NPTS + n], az = x1[2 * NPTS + n];
  float s1 = ax * ax + ay * ay + az * az;
  float bd[K_NN]; int bi[K_NN];
#pragma unroll
  for (int i = 0; i < K_NN; i++) { bd[i] = 3.4e38f; bi[i] = 0; }
  for (int m = 0; m < NPTS; m++) {
    float bx = x2[0 * NPTS + m], by = x2[1 * NPTS + m], bz = x2[2 * NPTS + m];
    float s2 = bx * bx + by * by + bz * bz;
    float d = fmaxf(-2.f * (ax * bx + ay * by + az * bz) + s1 + s2, 0.f);
    if (d < bd[K_NN - 1]) {                 // insertion keeps earliest index on ties
      int j = K_NN - 1;
      while (j > 0 && bd[j - 1] > d) { bd[j] = bd[j - 1]; bi[j] = bi[j - 1]; j--; }
      bd[j] = d; bi[j] = m;
    }
  }
  for (int i = 0; i < K_NN; i++) knn_idx[n * K_NN + i] = bi[i];
}

// ----------------------------------------------- std(diff): stage 1 --------
__global__ void stats_kernel(const float* __restrict__ p1, const float* __restrict__ x1,
                             const float* __restrict__ p2, const float* __restrict__ x2,
                             const int* __restrict__ idx, double* __restrict__ partials) {
  const long long TOT = (long long)MROWS * CCH;
  double s = 0.0, q = 0.0;
  for (long long g = (long long)blockIdx.x * blockDim.x + threadIdx.x; g < TOT;
       g += (long long)gridDim.x * blockDim.x) {
    int r = (int)(g / CCH), c = (int)(g % CCH);
    int n = r >> 4, m = idx[r];
    float g2v = (c < DCH) ? p2[c * NPTS + m] : x2[(c - DCH) * NPTS + m];
    float g1v = (c < DCH) ? p1[c * NPTS + n] : x1[(c - DCH) * NPTS + n];
    double d = (double)(g2v - g1v);
    s += d; q += d * d;
  }
  __shared__ double ss[256], sq[256];
  ss[threadIdx.x] = s; sq[threadIdx.x] = q; __syncthreads();
  for (int o = 128; o > 0; o >>= 1) {
    if ((int)threadIdx.x < o) { ss[threadIdx.x] += ss[threadIdx.x + o]; sq[threadIdx.x] += sq[threadIdx.x + o]; }
    __syncthreads();
  }
  if (threadIdx.x == 0) { partials[2 * blockIdx.x] = ss[0]; partials[2 * blockIdx.x + 1] = sq[0]; }
}

// ----------------------------------------------- std(diff): stage 2 --------
__global__ void std_kernel(const double* __restrict__ partials, int nblk, float* __restrict__ sinv) {
  double s = 0.0, q = 0.0;
  for (int i = threadIdx.x; i < nblk; i += blockDim.x) { s += partials[2 * i]; q += partials[2 * i + 1]; }
  __shared__ double ss[256], sq[256];
  ss[threadIdx.x] = s; sq[threadIdx.x] = q; __syncthreads();
  for (int o = 128; o > 0; o >>= 1) {
    if ((int)threadIdx.x < o) { ss[threadIdx.x] += ss[threadIdx.x + o]; sq[threadIdx.x] += sq[threadIdx.x + o]; }
    __syncthreads();
  }
  if (threadIdx.x == 0) {
    double M = (double)MROWS * (double)CCH;
    double var = (sq[0] - ss[0] * ss[0] / M) / (M - 1.0);   // torch.std unbiased
    double sd = sqrt(var > 0.0 ? var : 0.0);
    *sinv = (float)(1.0 / (sd + 1e-5));
  }
}

// -------------------- fold alpha/s into weights; transpose to bf16 ---------
__global__ void fold_kernel(const float* __restrict__ w0, const float* __restrict__ b0,
                            const float* __restrict__ w1, const float* __restrict__ alpha,
                            const float* __restrict__ beta, const float* __restrict__ sinvp,
                            __hip_bfloat16* __restrict__ w0g1t, __hip_bfloat16* __restrict__ w0g2t,
                            __hip_bfloat16* __restrict__ w1t, float* __restrict__ b0eff) {
  int o = blockIdx.x * blockDim.x + threadIdx.x;
  if (o >= NOUT) return;
  float sinv = *sinvp;
  float bacc = 0.f;
  for (int c = 0; c < KP; c++) {
    float g1w = 0.f, g2w = 0.f;
    if (c < CCH) {
      float wp3 = w0[(2 * CCH + c) * NOUT + o];
      float a = alpha[c] * sinv;
      g1w = w0[c * NOUT + o] - a * wp3;
      g2w = w0[(CCH + c) * NOUT + o] + a * wp3;
      bacc += beta[c] * wp3;
    }
    w0g1t[(size_t)o * KP + c] = __float2bfloat16(g1w);
    w0g2t[(size_t)o * KP + c] = __float2bfloat16(g2w);
  }
  b0eff[o] = b0[o] + bacc;
  for (int c = 0; c < NOUT; c++) w1t[(size_t)o * NOUT + c] = __float2bfloat16(w1[c * NOUT + o]);
}

// ------------------------------------------------ pack activations --------
__global__ void pack_g1_kernel(const float* __restrict__ p1, const float* __restrict__ x1,
                               __hip_bfloat16* __restrict__ g1bf) {
  int g = blockIdx.x * blockDim.x + threadIdx.x;   // NPTS*KP threads
  int n = g / KP, c = g % KP;
  float v = (c < DCH) ? p1[c * NPTS + n] : ((c < CCH) ? x1[(c - DCH) * NPTS + n] : 0.f);
  g1bf[g] = __float2bfloat16(v);
}

__global__ void pack_g2_kernel(const float* __restrict__ p2, const float* __restrict__ x2,
                               const int* __restrict__ idx, __hip_bfloat16* __restrict__ g2bf) {
  int g = blockIdx.x * blockDim.x + threadIdx.x;   // MROWS*KP threads
  int r = g / KP, c = g % KP;
  int m = idx[r];
  float v = (c < DCH) ? p2[c * NPTS + m] : ((c < CCH) ? x2[(c - DCH) * NPTS + m] : 0.f);
  g2bf[g] = __float2bfloat16(v);
}

// ------------------------------------------------ WeightNet (tiny MLP) ----
__global__ void weightnet_kernel(const float* __restrict__ x1, const float* __restrict__ x2,
                                 const int* __restrict__ idx,
                                 const float* __restrict__ w0, const float* __restrict__ b0,
                                 const float* __restrict__ w1, const float* __restrict__ b1,
                                 const float* __restrict__ w2, const float* __restrict__ b2,
                                 __hip_bfloat16* __restrict__ wbf) {
  int r = blockIdx.x * 8 + (threadIdx.x >> 5);   // one wave32 per (n,k) row
  int lane = threadIdx.x & 31;
  int n = r >> 4, m = idx[r];
  float dx = x2[0 * NPTS + m] - x1[0 * NPTS + n];
  float dy = x2[1 * NPTS + m] - x1[1 * NPTS + n];
  float dz = x2[2 * NPTS + m] - x1[2 * NPTS + n];
  float h1[8], h2[8];
#pragma unroll
  for (int o = 0; o < 8; o++) {
    float a = dx * w0[0 * 8 + o] + dy * w0[1 * 8 + o] + dz * w0[2 * 8 + o] + b0[o];
    h1[o] = fmaxf(a, 0.f);
  }
#pragma unroll
  for (int o = 0; o < 8; o++) {
    float a = b1[o];
#pragma unroll
    for (int c = 0; c < 8; c++) a += h1[c] * w1[c * 8 + o];
    h2[o] = fmaxf(a, 0.f);
  }
#pragma unroll
  for (int i = 0; i < 16; i++) {
    int o = i * 32 + lane;
    float a = b2[o];
#pragma unroll
    for (int c = 0; c < 8; c++) a += h2[c] * w2[c * NOUT + o];
    wbf[(size_t)r * NOUT + o] = __float2bfloat16(fmaxf(a, 0.f));
  }
}

// --------------------------------------------------- bf16 WMMA GEMM ------
// C[M x 512] = act(A[M x Kdim] * Wt^T + bias + rowBias), Wt stored [512 x Kdim].
// Block = 256 threads = 8 waves; wave w computes rows [w*16, w*16+16) of a
// 128-row block across a 64-col strip (4 WMMA accumulators, A frag reused).
// Tiles staged by async global->LDS DMA (double buffered): the transfer for
// step k+1 overlaps the WMMA work for step k; one s_wait_asynccnt + barrier
// per step. All four B fragments are preloaded so the ds_loads form one
// clause and overlap WMMA issue instead of a dscnt-0 wait per WMMA.
// LDS rows padded to 80B to spread the 64 banks on frag gathers.
#define A_TILE (128 * 80)
#define W_TILE (64 * 80)

template <int ACT_LEAKY, int OUT_BF16, int HAS_BIAS, int HAS_ROWBIAS>
__global__ __launch_bounds__(256) void gemm_kernel(
    const __hip_bfloat16* __restrict__ A, int Kdim,
    const __hip_bfloat16* __restrict__ Wt,
    const float* __restrict__ bias, const float* __restrict__ rowBias, int rowShift,
    void* __restrict__ outp) {
  __shared__ __align__(16) unsigned char As[2][A_TILE];
  __shared__ __align__(16) unsigned char Ws[2][W_TILE];
  const int tid = threadIdx.x;
  const int wave = tid >> 5, lane = tid & 31;
  const int half = lane >> 4, col = lane & 15;
  const int mBlk = blockIdx.x * 128;
  const int nBlk = blockIdx.y * 64;

  v8f acc[4];
#pragma unroll
  for (int j = 0; j < 4; j++) { v8f z = {}; acc[j] = z; }

  // cooperative async-DMA descriptors (per lane)
  const int am = tid >> 1, ah = tid & 1;   // A: 128 rows x 2 chunks of 16 bf16
  const int wo = tid >> 2, wq = tid & 3;   // Wt: 64 rows x 4 chunks of 8 bf16
  const __hip_bfloat16* Aptr = A + (size_t)(mBlk + am) * Kdim + ah * 16;
  const __hip_bfloat16* Wptr = Wt + (size_t)(nBlk + wo) * Kdim + wq * 8;
  // low 32 bits of a flat shared-aperture address = workgroup-relative LDS byte addr
  const unsigned ldsA = (unsigned)(unsigned long long)(const void*)&As[0][0] + am * 80 + ah * 32;
  const unsigned ldsW = (unsigned)(unsigned long long)(const void*)&Ws[0][0] + wo * 80 + wq * 16;

  auto issue = [&](int k0, int db) {
    unsigned la = ldsA + db * A_TILE;
    unsigned long long ga = (unsigned long long)(const void*)(Aptr + k0);
    // inst offset applies to BOTH lds and global sides (ISA 10.x async pseudocode)
    asm volatile("global_load_async_to_lds_b128 %0, %1, off" :: "v"(la), "v"(ga) : "memory");
    asm volatile("global_load_async_to_lds_b128 %0, %1, off offset:16" :: "v"(la), "v"(ga) : "memory");
    unsigned lw = ldsW + db * W_TILE;
    unsigned long long gw = (unsigned long long)(const void*)(Wptr + k0);
    asm volatile("global_load_async_to_lds_b128 %0, %1, off" :: "v"(lw), "v"(gw) : "memory");
  };

  issue(0, 0);
  asm volatile("s_wait_asynccnt 0x0" ::: "memory");
  __syncthreads();

  for (int k0 = 0; k0 < Kdim; k0 += 32) {
    const int db = (k0 >> 5) & 1;
    const bool more = (k0 + 32 < Kdim);
    if (more) issue(k0 + 32, db ^ 1);   // overlap next-tile DMA with this tile's WMMAs

    // A fragment (16x32 bf16), contiguous-K layout (same K order as B): lane L (row=L%16, half=L/16),
    // VGPR v holds K = half*16 + 2v .. +1
    FragAB a;
    const unsigned char* arow = &As[db][(wave * 16 + col) * 80];
#pragma unroll
    for (int v = 0; v < 8; v++) {
      int kk = half * 16 + 2 * v;
      a.u[v] = *(const unsigned int*)(arow + kk * 2);
    }
    // B fragments (32x16): lane L (col=L%16, half=L/16), VGPR v holds
    // K = half*16 + 2v .. +1  -> 16 consecutive K per lane (2 x uint4).
    // Preload all four so ds_loads clause together and overlap WMMA issue.
    FragAB b[4];
#pragma unroll
    for (int j = 0; j < 4; j++) {
      const uint4* bp = (const uint4*)(&Ws[db][(j * 16 + col) * 80 + half * 32]);
      uint4 b0v = bp[0], b1v = bp[1];
      b[j].u[0] = b0v.x; b[j].u[1] = b0v.y; b[j].u[2] = b0v.z; b[j].u[3] = b0v.w;
      b[j].u[4] = b1v.x; b[j].u[5] = b1v.y; b[j].u[6] = b1v.z; b[j].u[7] = b1v.w;
    }
#pragma unroll
    for (int j = 0; j < 4; j++) {
      acc[j] = __builtin_amdgcn_wmma_f32_16x16x32_bf16(
          false, a.v, false, b[j].v, (short)0, acc[j], false, false);
    }

    if (more) {
      asm volatile("s_wait_asynccnt 0x0" ::: "memory");  // next tile landed
      __syncthreads();                                   // everyone done with this tile
    }
  }

  // epilogue: C layout lane L -> col=L%16, VGPR i -> row = i + 8*(L/16).
  // A 16-row wave tile spans exactly one (R >> rowShift) group, so the
  // row-bias is one load per output column.
#pragma unroll
  for (int j = 0; j < 4; j++) {
    int O = nBlk + j * 16 + col;
    float bb = 0.f;
    if (HAS_BIAS)    bb += bias[O];
    if (HAS_ROWBIAS) bb += rowBias[(size_t)((mBlk + wave * 16) >> rowShift) * NOUT + O];
#pragma unroll
    for (int i = 0; i < 8; i++) {
      int R = mBlk + wave * 16 + half * 8 + i;
      float v = acc[j][i] + bb;
      if (ACT_LEAKY) v = (v > 0.f) ? v : 0.1f * v;
      if (OUT_BF16)
        ((__hip_bfloat16*)outp)[(size_t)R * NOUT + O] = __float2bfloat16(v);
      else
        ((float*)outp)[(size_t)R * NOUT + O] = v;
    }
  }
}

// ---------------------------------------------- final w*h K-sum ----------
__global__ void reduce_kernel(const __hip_bfloat16* __restrict__ wbf,
                              const __hip_bfloat16* __restrict__ h2,
                              float* __restrict__ out) {
  int n = blockIdx.x;        // 4096 blocks
  int c = threadIdx.x;       // 512 threads
  float acc = 0.f;
#pragma unroll
  for (int k = 0; k < K_NN; k++) {
    size_t off = ((size_t)(n * K_NN + k)) * NOUT + c;
    acc += __bfloat162float(wbf[off]) * __bfloat162float(h2[off]);
  }
  out[(size_t)c * NPTS + n] = acc;   // [B,512,N]
}

// ---------------------------------------------------------------------------
extern "C" void kernel_launch(void* const* d_in, const int* in_sizes, int n_in,
                              void* d_out, int out_size, void* d_ws, size_t ws_size,
                              hipStream_t stream) {
  (void)in_sizes; (void)n_in; (void)out_size; (void)ws_size;
  // inputs (reference uses only the underscored tensors + params)
  const float* xyz1_   = (const float*)d_in[4];
  const float* xyz2_   = (const float*)d_in[5];
  const float* points1 = (const float*)d_in[6];
  const float* points2 = (const float*)d_in[7];
  const float* w0      = (const float*)d_in[8];
  const float* b0      = (const float*)d_in[9];
  const float* w1      = (const float*)d_in[10];
  const float* b1      = (const float*)d_in[11];
  const float* wn_w0   = (const float*)d_in[12];
  const float* wn_b0   = (const float*)d_in[13];
  const float* wn_w1   = (const float*)d_in[14];
  const float* wn_b1   = (const float*)d_in[15];
  const float* wn_w2   = (const float*)d_in[16];
  const float* wn_b2   = (const float*)d_in[17];
  const float* alpha   = (const float*)d_in[18];
  const float* beta    = (const float*)d_in[19];

  // workspace carve-out (~287 MB)
  char* base = (char*)d_ws;
  size_t off = 0;
  auto take = [&](size_t bytes) -> void* {
    void* p = base + off;
    off = (off + bytes + 255) & ~(size_t)255;
    return p;
  };
  int*            idx      = (int*)take((size_t)MROWS * 4);
  double*         partials = (double*)take((size_t)4096 * 2 * 8);
  float*          sinv     = (float*)take(256);
  __hip_bfloat16* w0g1t    = (__hip_bfloat16*)take((size_t)NOUT * KP * 2);
  __hip_bfloat16* w0g2t    = (__hip_bfloat16*)take((size_t)NOUT * KP * 2);
  __hip_bfloat16* w1t      = (__hip_bfloat16*)take((size_t)NOUT * NOUT * 2);
  float*          b0eff    = (float*)take((size_t)NOUT * 4);
  __hip_bfloat16* g1bf     = (__hip_bfloat16*)take((size_t)NPTS * KP * 2);
  __hip_bfloat16* g2bf     = (__hip_bfloat16*)take((size_t)MROWS * KP * 2);
  __hip_bfloat16* wbf      = (__hip_bfloat16*)take((size_t)MROWS * NOUT * 2);
  float*          T1       = (float*)take((size_t)NPTS * NOUT * 4);
  __hip_bfloat16* H1       = (__hip_bfloat16*)take((size_t)MROWS * NOUT * 2);
  __hip_bfloat16* H2       = (__hip_bfloat16*)take((size_t)MROWS * NOUT * 2);

  knn_kernel<<<NPTS / 256, 256, 0, stream>>>(xyz1_, xyz2_, idx);
  stats_kernel<<<4096, 256, 0, stream>>>(points1, xyz1_, points2, xyz2_, idx, partials);
  std_kernel<<<1, 256, 0, stream>>>(partials, 4096, sinv);
  fold_kernel<<<2, 256, 0, stream>>>(w0, b0, w1, alpha, beta, sinv, w0g1t, w0g2t, w1t, b0eff);
  pack_g1_kernel<<<(NPTS * KP) / 256, 256, 0, stream>>>(points1, xyz1_, g1bf);
  pack_g2_kernel<<<(MROWS * KP) / 256, 256, 0, stream>>>(points2, xyz2_, idx, g2bf);
  weightnet_kernel<<<MROWS / 8, 256, 0, stream>>>(xyz1_, xyz2_, idx, wn_w0, wn_b0,
                                                  wn_w1, wn_b1, wn_w2, wn_b2, wbf);
  // T1[n, o] = g1 . W0g1 + b0eff   (per-point term, fp32)
  gemm_kernel<0, 0, 1, 0><<<dim3(NPTS / 128, NOUT / 64), 256, 0, stream>>>(
      g1bf, KP, w0g1t, b0eff, nullptr, 0, (void*)T1);
  // H1 = leaky(g2 . W0g2 + T1[row>>4])
  gemm_kernel<1, 1, 0, 1><<<dim3(MROWS / 128, NOUT / 64), 256, 0, stream>>>(
      g2bf, KP, w0g2t, nullptr, T1, 4, (void*)H1);
  // H2 = leaky(H1 . W1 + b1)
  gemm_kernel<1, 1, 1, 0><<<dim3(MROWS / 128, NOUT / 64), 256, 0, stream>>>(
      H1, NOUT, w1t, b1, nullptr, 0, (void*)H2);
  reduce_kernel<<<NPTS, 512, 0, stream>>>(wbf, H2, (float*)d_out);
}
